// SetAbstraction_64510408786139
// MI455X (gfx1250) — matrix operand order repaired and checked by reading, not compiled, not run on hardware
//
#include <hip/hip_runtime.h>
#include <hip/hip_bf16.h>

// Problem constants (fixed by setup_inputs in the reference)
#define BB   4
#define NN   8192
#define CC   64
#define MM   2048          // N / STRIDE
#define KK   32            // NSAMPLE
#define R2   0.01f         // RADIUS^2
#define EPSB 1e-5f
#define KSTR1 104          // xt column stride (halves): 96 rows + pad, mult of 8
#define KSTR2 72           // o1t column stride (halves): 64 rows + pad, mult of 8

typedef __attribute__((ext_vector_type(16))) _Float16 v16h;
typedef __attribute__((ext_vector_type(8)))  _Float16 v8h;
typedef __attribute__((ext_vector_type(8)))  float    v8f;

// ---------------------------------------------------------------------------
// WMMA fragment helpers (wave32; A/B use this code's contiguous k order, C/D per CDNA5 ISA 7.12.2)
// A (16x32 f16, MxK): lane L: m=L&15, g=L>>4; slot h -> k = 16g + h
// B (32x16 f16, KxN): lane L: n=L&15, g=L>>4; slot h -> k = 16g + h
// C/D (16x16 f32):    lane L: n=L&15, g=L>>4; VGPR r -> m = r + 8g
// ---------------------------------------------------------------------------

// A fragment from pre-packed (lane-ordered) weight buffer: one 32B load/lane.
__device__ __forceinline__ v16h load_frag_a_pre(const _Float16* __restrict__ wf,
                                                int fragIdx) {
    int lane = threadIdx.x & 31;
    return *(const v16h*)(wf + ((size_t)fragIdx * 32 + lane) * 16);
}

// B fragment from column-major LDS tile: lane's 16 halves are contiguous in k.
__device__ __forceinline__ v16h load_frag_b_ldsT(const _Float16* lds,
                                                 int k0, int col0, int kstr) {
    int lane = threadIdx.x & 31;
    int n = lane & 15, g = lane >> 4;
    const _Float16* src = lds + (col0 + n) * kstr + k0 + g * 16;
    v8h lo = *(const v8h*)(src);
    v8h hi = *(const v8h*)(src + 8);
    v16h b;
#pragma unroll
    for (int h = 0; h < 8; ++h) { b[h] = lo[h]; b[h + 8] = hi[h]; }
    return b;
}

// ---------------------------------------------------------------------------
// Kernel 0a: zero the BN stat accumulators (ws is not re-poisoned per replay)
// ---------------------------------------------------------------------------
__global__ void zero_stats_kernel(float* __restrict__ stats) {
    int t = threadIdx.x;
    if (t < 384) stats[t] = 0.0f;
}

// ---------------------------------------------------------------------------
// Kernel 0b: pack w1/w2 into WMMA-lane-ordered f16 fragments.
// wf1: [rt(4) x ks(3)] frags, wf2: [rt(8) x ks(2)] frags; each 32 lanes x 16 h
// ---------------------------------------------------------------------------
__global__ __launch_bounds__(256) void prep_weights_kernel(const float* __restrict__ w1,
                                                           const float* __restrict__ w2,
                                                           _Float16* __restrict__ wf1,
                                                           _Float16* __restrict__ wf2) {
    int t = blockIdx.x * 256 + threadIdx.x;
    if (t < 4 * 3 * 32 * 16) {                       // 6144
        int h = t & 15, lane = (t >> 4) & 31, frag = t >> 9;
        int rt = frag / 3, ksi = frag % 3;
        int m = lane & 15, g = lane >> 4;
        int k = ksi * 32 + g * 16 + h;
        float v = (k < 67) ? w1[(rt * 16 + m) * 67 + k] : 0.0f;
        wf1[t] = (_Float16)v;
    }
    if (t < 8 * 2 * 32 * 16) {                       // 8192
        int h = t & 15, lane = (t >> 4) & 31, frag = t >> 9;
        int rt = frag >> 1, ksi = frag & 1;
        int m = lane & 15, g = lane >> 4;
        int k = ksi * 32 + g * 16 + h;
        wf2[t] = (_Float16)w2[(rt * 16 + m) * 64 + k];
    }
}

// ---------------------------------------------------------------------------
// Kernel 1: furthest point sampling. One block per batch, LDS dmin array.
// Tie-break = lowest index (matches jnp.argmax).
// ---------------------------------------------------------------------------
__global__ __launch_bounds__(256) void fps_kernel(const float* __restrict__ p,
                                                  int* __restrict__ idx_ws,
                                                  int* __restrict__ idx_out) {
    __shared__ float dmin[NN];
    __shared__ float rv[256];
    __shared__ int   ri[256];
    __shared__ int   sel;
    const int b = blockIdx.x, t = threadIdx.x;
    const float* pb = p + (size_t)b * NN * 3;
    for (int i = t; i < NN; i += 256) dmin[i] = 3.4e38f;
    if (t == 0) { sel = 0; idx_ws[b * MM] = 0; idx_out[b * MM] = 0; }
    __syncthreads();
    int last = 0;
    for (int it = 1; it < MM; ++it) {
        float lx = pb[last * 3 + 0], ly = pb[last * 3 + 1], lz = pb[last * 3 + 2];
        float bestv = -1.0f; int besti = NN;
        for (int i = t; i < NN; i += 256) {
            float dx = pb[i * 3 + 0] - lx;
            float dy = pb[i * 3 + 1] - ly;
            float dz = pb[i * 3 + 2] - lz;
            float d  = dx * dx + dy * dy + dz * dz;
            float dm = fminf(dmin[i], d);
            dmin[i] = dm;
            if (dm > bestv) { bestv = dm; besti = i; }
        }
        rv[t] = bestv; ri[t] = besti;
        __syncthreads();
        for (int s = 128; s > 0; s >>= 1) {
            if (t < s) {
                float v2 = rv[t + s]; int i2 = ri[t + s];
                if (v2 > rv[t] || (v2 == rv[t] && i2 < ri[t])) { rv[t] = v2; ri[t] = i2; }
            }
            __syncthreads();
        }
        if (t == 0) { sel = ri[0]; idx_ws[b * MM + it] = sel; idx_out[b * MM + it] = sel; }
        __syncthreads();
        last = sel;
    }
}

// ---------------------------------------------------------------------------
// Kernel 2: gather sampled points -> ws copy + d_out new_p
// ---------------------------------------------------------------------------
__global__ __launch_bounds__(256) void gather_newp_kernel(const float* __restrict__ p,
                                                          const int* __restrict__ idx,
                                                          float* __restrict__ newp_ws,
                                                          float* __restrict__ out_newp) {
    int i = blockIdx.x * 256 + threadIdx.x;           // over B*M
    if (i >= BB * MM) return;
    int b = i / MM;
    int j = idx[i];
    const float* s = p + ((size_t)b * NN + j) * 3;
    float x = s[0], y = s[1], z = s[2];
    newp_ws[i * 3 + 0] = x; newp_ws[i * 3 + 1] = y; newp_ws[i * 3 + 2] = z;
    out_newp[i * 3 + 0] = x; out_newp[i * 3 + 1] = y; out_newp[i * 3 + 2] = z;
}

// ---------------------------------------------------------------------------
// Kernel 3: residual skip branch identity[b,o,m] = w_skip @ f[:, idx] + b_skip
// (134 MFLOP total; f is L2-resident, plain VALU is fine)
// ---------------------------------------------------------------------------
__global__ __launch_bounds__(256) void identity_kernel(const float* __restrict__ f,
                                                       const int* __restrict__ idx,
                                                       const float* __restrict__ wsk,
                                                       const float* __restrict__ bsk,
                                                       float* __restrict__ ident) {
    int i = blockIdx.x * 256 + threadIdx.x;           // over B*128*M
    if (i >= BB * 128 * MM) return;
    int m = i & (MM - 1);
    int o = (i / MM) & 127;
    int b = i / (128 * MM);
    int j = idx[b * MM + m];
    const float* fb = f + (size_t)b * CC * NN + j;
    const float* w  = wsk + o * CC;
    float acc = bsk[o];
#pragma unroll 8
    for (int c = 0; c < CC; ++c) acc += w[c] * fb[(size_t)c * NN];
    ident[i] = acc;
}

// ---------------------------------------------------------------------------
// Kernel 4: ball query — one wave32 per query. CUDA ballquery semantics:
// first K indices (ascending) with d2 < r2, padded with first found (0 if none)
// ---------------------------------------------------------------------------
__global__ __launch_bounds__(256) void ballquery_kernel(const float* __restrict__ p,
                                                        const float* __restrict__ newp,
                                                        int* __restrict__ nidx) {
    int wave = threadIdx.x >> 5, lane = threadIdx.x & 31;
    int q = blockIdx.x * 8 + wave;                    // global query in [0, B*M)
    int b = q / MM;
    const float* pb = p + (size_t)b * NN * 3;
    float qx = newp[q * 3 + 0], qy = newp[q * 3 + 1], qz = newp[q * 3 + 2];
    int* out = nidx + (size_t)q * KK;
    int cnt = 0, first = 0;
    for (int base = 0; base < NN && cnt < KK; base += 32) {
        int j = base + lane;
        float dx = pb[j * 3 + 0] - qx;
        float dy = pb[j * 3 + 1] - qy;
        float dz = pb[j * 3 + 2] - qz;
        bool inr = (dx * dx + dy * dy + dz * dz) < R2;
        unsigned mask = __builtin_amdgcn_ballot_w32(inr);
        if (cnt == 0 && mask) first = base + __builtin_ctz(mask);
        int navail = KK - cnt;
        if (inr) {
            int rank = __builtin_popcount(mask & ((1u << lane) - 1u));
            if (rank < navail) out[cnt + rank] = j;
        }
        cnt += __builtin_popcount(mask);
        if (cnt > KK) cnt = KK;
    }
    if (lane >= cnt) out[lane] = first;               // pad (KK == 32 == wave)
}

// ---------------------------------------------------------------------------
// Kernel 5 (template): fused gather + conv1[+bn1+relu+conv2][+bn2+maxK+skip]
// MODE 0: conv1 raw -> BN1 stats
// MODE 1: conv1+bn1+relu -> conv2 raw -> BN2 stats
// MODE 2: full path -> bn2 -> max over K -> +identity -> relu -> fout
// 4 query points (=128 GEMM columns) per block, 256 threads = 8 waves.
// LDS tiles are COLUMN-MAJOR so B-fragments load as contiguous 16B chunks.
// stats layout: [s1sum 64][s1sq 64][s2sum 128][s2sq 128]
// bnp   layout: [scale1 64][shift1 64][scale2 128][shift2 128]
// ---------------------------------------------------------------------------
template <int MODE>
__global__ __launch_bounds__(256) void conv_kernel(const float* __restrict__ p,
                                                   const float* __restrict__ f,
                                                   const _Float16* __restrict__ wf1,
                                                   const _Float16* __restrict__ wf2,
                                                   const float* __restrict__ newp,
                                                   const int* __restrict__ nidx,
                                                   float* __restrict__ stats,
                                                   const float* __restrict__ bnp,
                                                   const float* __restrict__ ident,
                                                   float* __restrict__ fout) {
    __shared__ _Float16 xt[128 * KSTR1];              // X tile, col-major [c][k]
    __shared__ _Float16 o1t[128 * KSTR2];             // conv1 tile, col-major
    __shared__ float    sacc[256];                    // per-block BN stat scratch

    const int tid  = threadIdx.x;
    const int gq0  = blockIdx.x * 4;                  // first of 4 queries
    const int wv   = tid >> 5;
    const int lane = tid & 31;
    const int ln   = lane & 15, lg = lane >> 4;

    if (MODE == 0 || MODE == 1) sacc[tid] = 0.0f;

    // ---- build X tile: k 0..2 = dp, 3..66 = gathered f, 67..95 = zero ----
    for (int i = tid; i < 128 * 96; i += 256) {
        int c = i / 96, k = i % 96;
        int q = gq0 + (c >> 5);
        int b = q / MM;
        int j = nidx[(size_t)q * KK + (c & 31)];
        float v;
        if (k < 3)       v = p[((size_t)b * NN + j) * 3 + k] - newp[(size_t)q * 3 + k];
        else if (k < 67) v = f[((size_t)b * CC + (k - 3)) * NN + j];
        else             v = 0.0f;
        xt[c * KSTR1 + k] = (_Float16)v;
    }
    __syncthreads();

    // ---- conv1: 4 row-tiles x 8 col-tiles, K = 96 (3 WMMA steps) ----
    for (int jj = 0; jj < 4; ++jj) {
        int job = wv * 4 + jj;                        // 0..31
        int rt = job >> 3, ct = job & 7;
        v8f acc = {};
#pragma unroll
        for (int ksi = 0; ksi < 3; ++ksi) {
            v16h a   = load_frag_a_pre(wf1, rt * 3 + ksi);
            v16h bfr = load_frag_b_ldsT(xt, ksi * 32, ct * 16, KSTR1);
            acc = __builtin_amdgcn_wmma_f32_16x16x32_f16(
                false, a, false, bfr, (short)0, acc, false, false);
        }
        if (MODE == 0) {
#pragma unroll
            for (int r = 0; r < 8; ++r) {
                int ch = rt * 16 + r + 8 * lg;
                float v = acc[r];
                atomicAdd(&sacc[ch], v);
                atomicAdd(&sacc[64 + ch], v * v);
            }
        } else {
#pragma unroll
            for (int r = 0; r < 8; ++r) {
                int ch = rt * 16 + r + 8 * lg;
                float v = fmaxf(acc[r] * bnp[ch] + bnp[64 + ch], 0.0f);
                o1t[(ct * 16 + ln) * KSTR2 + ch] = (_Float16)v;
            }
        }
    }
    __syncthreads();

    // ---- conv2: 8 row-tiles x 4 queries (2 col-tiles each), K = 64 ----
    if (MODE >= 1) {
        for (int jj = 0; jj < 4; ++jj) {
            int job = wv * 4 + jj;                    // 0..31
            int rt = job >> 2;                        // 0..7
            int q  = job & 3;
            float mx[8];
#pragma unroll
            for (int r = 0; r < 8; ++r) mx[r] = -3.4e38f;
#pragma unroll
            for (int half = 0; half < 2; ++half) {
                int ct = q * 2 + half;
                v8f acc = {};
#pragma unroll
                for (int ksi = 0; ksi < 2; ++ksi) {
                    v16h a   = load_frag_a_pre(wf2, rt * 2 + ksi);
                    v16h bfr = load_frag_b_ldsT(o1t, ksi * 32, ct * 16, KSTR2);
                    acc = __builtin_amdgcn_wmma_f32_16x16x32_f16(
                        false, a, false, bfr, (short)0, acc, false, false);
                }
                if (MODE == 1) {
#pragma unroll
                    for (int r = 0; r < 8; ++r) {
                        int ch = rt * 16 + r + 8 * lg;
                        float v = acc[r];
                        atomicAdd(&sacc[ch], v);
                        atomicAdd(&sacc[128 + ch], v * v);
                    }
                } else {
#pragma unroll
                    for (int r = 0; r < 8; ++r) {
                        int ch = rt * 16 + r + 8 * lg;
                        float v = acc[r] * bnp[128 + ch] + bnp[256 + ch];
                        mx[r] = fmaxf(mx[r], v);
                    }
                }
            }
            if (MODE == 2) {
                // butterfly max over the 16 column lanes in each half-wave
#pragma unroll
                for (int r = 0; r < 8; ++r) {
                    float v = mx[r];
                    v = fmaxf(v, __shfl_xor(v, 1, 32));
                    v = fmaxf(v, __shfl_xor(v, 2, 32));
                    v = fmaxf(v, __shfl_xor(v, 4, 32));
                    v = fmaxf(v, __shfl_xor(v, 8, 32));
                    mx[r] = v;
                }
                if (ln == 0) {                        // lanes 0 and 16 write
                    int gq = gq0 + q;
                    int b = gq / MM, m = gq & (MM - 1);
#pragma unroll
                    for (int r = 0; r < 8; ++r) {
                        int ch = rt * 16 + r + 8 * lg;
                        size_t o = ((size_t)b * 128 + ch) * MM + m;
                        fout[o] = fmaxf(mx[r] + ident[o], 0.0f);
                    }
                }
            }
        }
    }

    // ---- flush per-block BN stats ----
    if (MODE == 0) {
        __syncthreads();
        if (tid < 128) atomicAdd(&stats[tid], sacc[tid]);
    }
    if (MODE == 1) {
        __syncthreads();
        if (tid < 256) atomicAdd(&stats[128 + tid], sacc[tid]);
    }
}

// ---------------------------------------------------------------------------
// Kernels 6/8: fold BN stats into per-channel scale/shift
// ---------------------------------------------------------------------------
__global__ void finalize1_kernel(const float* __restrict__ stats,
                                 const float* __restrict__ g,
                                 const float* __restrict__ b,
                                 float* __restrict__ bnp) {
    int c = threadIdx.x;                              // 64
    float inv  = 1.0f / (float)(BB * MM * KK);
    float mean = stats[c] * inv;
    float var  = stats[64 + c] * inv - mean * mean;
    float sc   = g[c] * rsqrtf(var + EPSB);
    bnp[c]      = sc;
    bnp[64 + c] = b[c] - mean * sc;
}

__global__ void finalize2_kernel(const float* __restrict__ stats,
                                 const float* __restrict__ g,
                                 const float* __restrict__ b,
                                 float* __restrict__ bnp) {
    int c = threadIdx.x;                              // 128
    float inv  = 1.0f / (float)(BB * MM * KK);
    float mean = stats[128 + c] * inv;
    float var  = stats[256 + c] * inv - mean * mean;
    float sc   = g[c] * rsqrtf(var + EPSB);
    bnp[128 + c] = sc;
    bnp[256 + c] = b[c] - mean * sc;
}

// ---------------------------------------------------------------------------
// Launcher. d_out = [new_p (B*M*3)][fout (B*128*M)][idx (B*M, int bits)]
// ws layout (bytes): idx@0, newp@32768, nidx@131072, ident@1179648,
//   stats@5373952 (384f), bnp@5375488 (384f), wf1@5377024 (6144h),
//   wf2@5389312 (8192h); end ~5.41 MB
// ---------------------------------------------------------------------------
extern "C" void kernel_launch(void* const* d_in, const int* in_sizes, int n_in,
                              void* d_out, int out_size, void* d_ws, size_t ws_size,
                              hipStream_t stream) {
    const float* p   = (const float*)d_in[0];
    const float* f   = (const float*)d_in[1];
    const float* w1  = (const float*)d_in[2];
    const float* g1  = (const float*)d_in[3];
    const float* b1  = (const float*)d_in[4];
    const float* w2  = (const float*)d_in[5];
    const float* g2  = (const float*)d_in[6];
    const float* b2  = (const float*)d_in[7];
    const float* wsk = (const float*)d_in[8];
    const float* bsk = (const float*)d_in[9];

    char*      ws    = (char*)d_ws;
    int*       idx   = (int*)(ws + 0);
    float*     newp  = (float*)(ws + 32768);
    int*       nidx  = (int*)(ws + 131072);
    float*     ident = (float*)(ws + 1179648);
    float*     stats = (float*)(ws + 5373952);
    float*     bnp   = (float*)(ws + 5375488);
    _Float16*  wf1   = (_Float16*)(ws + 5377024);
    _Float16*  wf2   = (_Float16*)(ws + 5389312);

    float* out_newp = (float*)d_out;
    float* out_fout = out_newp + (size_t)BB * MM * 3;
    int*   out_idx  = (int*)(out_fout + (size_t)BB * 128 * MM);

    zero_stats_kernel<<<1, 384, 0, stream>>>(stats);
    prep_weights_kernel<<<32, 256, 0, stream>>>(w1, w2, wf1, wf2);
    fps_kernel<<<BB, 256, 0, stream>>>(p, idx, out_idx);
    gather_newp_kernel<<<(BB * MM + 255) / 256, 256, 0, stream>>>(p, idx, newp, out_newp);
    identity_kernel<<<(BB * 128 * MM + 255) / 256, 256, 0, stream>>>(f, idx, wsk, bsk, ident);
    ballquery_kernel<<<BB * MM / 8, 256, 0, stream>>>(p, newp, nidx);

    conv_kernel<0><<<BB * MM / 4, 256, 0, stream>>>(p, f, wf1, wf2, newp, nidx,
                                                    stats, bnp, ident, out_fout);
    finalize1_kernel<<<1, 64, 0, stream>>>(stats, g1, b1, bnp);
    conv_kernel<1><<<BB * MM / 4, 256, 0, stream>>>(p, f, wf1, wf2, newp, nidx,
                                                    stats, bnp, ident, out_fout);
    finalize2_kernel<<<1, 128, 0, stream>>>(stats, g2, b2, bnp);
    conv_kernel<2><<<BB * MM / 4, 256, 0, stream>>>(p, f, wf1, wf2, newp, nidx,
                                                    stats, bnp, ident, out_fout);
}
